// NoiseEfficientNet_31533649887374
// MI455X (gfx1250) — matrix operand order repaired and checked by reading, not compiled, not run on hardware
//
#include <hip/hip_runtime.h>

// ---------------------------------------------------------------------------
// NoiseEfficientNet fused conv for MI455X (gfx1250, wave32)
//   main conv  : implicit GEMM, bf16x2-split V_WMMA_F32_16X16X32_BF16
//   extra path : collapsed to 9-case border table (conv of constant map)
//   staging    : pre-padded, pre-split bf16 image; async b128 -> LDS,
//                double-buffered against WMMA compute
// ---------------------------------------------------------------------------

typedef __attribute__((ext_vector_type(16))) __bf16 v16bf;
typedef __attribute__((ext_vector_type(8)))  float  v8f;
typedef __attribute__((ext_vector_type(4)))  int    v4i;

#define B_   32
#define CIN  96
#define COUT 144
#define H_   56
#define W_   56
#define HP   58   // padded height (zero border baked into packed image)
#define WP   58   // padded width

#define AS1 __attribute__((address_space(1)))
#define AS3 __attribute__((address_space(3)))

#if defined(__AMDGCN__) && __has_builtin(__builtin_amdgcn_global_load_async_to_lds_b128)
#define ASYNC_STAGE 1
#else
#define ASYNC_STAGE 0
#endif

// float -> bf16 round-to-nearest-even
static __device__ __forceinline__ unsigned short f2bf(float f) {
    union { float f; unsigned u; } x; x.f = f;
    unsigned r = x.u + 0x7fffu + ((x.u >> 16) & 1u);
    return (unsigned short)(r >> 16);
}
static __device__ __forceinline__ float bf2f(unsigned short h) {
    union { unsigned u; float f; } x; x.u = ((unsigned)h) << 16;
    return x.f;
}

union Frag32B { uint4 q[2]; v16bf v; };

static __device__ __forceinline__ v16bf load_frag(const unsigned short* p) {
    Frag32B u;
    u.q[0] = *(const uint4*)p;
    u.q[1] = *((const uint4*)p + 1);
    return u.v;
}

static __device__ __forceinline__ void wait_async_lds() {
#if ASYNC_STAGE
#if __has_builtin(__builtin_amdgcn_s_wait_asynccnt)
    __builtin_amdgcn_s_wait_asynccnt(0);
#else
    asm volatile("s_wait_asynccnt 0x0" ::: "memory");
#endif
#endif
}

// ---------------------------------------------------------------------------
// Pre-kernel 1: pack Wm (fp32 [144][96][3][3]) into WMMA A-fragments.
// Layout: whi/wlo[tap][chunk][mtile][lane][16 bf16]  (per-lane contiguous 32B)
// A layout (matches B frag): lane<16 holds K 0..15, lane>=16 K 16..31
// ---------------------------------------------------------------------------
__global__ void pack_w_kernel(const float* __restrict__ Wm,
                              unsigned short* __restrict__ whi,
                              unsigned short* __restrict__ wlo)
{
    int idx = blockIdx.x * 256 + threadIdx.x;
    if (idx >= 9 * 3 * 9 * 32 * 16) return;
    int j = idx & 15;
    int t = idx >> 4;
    int lane = t & 31;  t >>= 5;
    int mtile = t % 9;  t /= 9;
    int chunk = t % 3;
    int tap   = t / 3;

    int base = (lane >> 4) * 16;
    int k  = base + j;
    int ci = chunk * 32 + k;
    int co = mtile * 16 + (lane & 15);
    int kh = tap / 3, kw = tap % 3;

    float v = Wm[((co * CIN + ci) * 3 + kh) * 3 + kw];
    unsigned short hi = f2bf(v);
    whi[idx] = hi;
    wlo[idx] = f2bf(v - bf2f(hi));
}

// ---------------------------------------------------------------------------
// Pre-kernel 2: extra path + biases -> 9-case border table
//   ebias[b][co][rc*3+cc] = bm[co] + bx[co]
//       + sum_f extra[b,co,f] * sum_{kh in R(rc), kw in C(cc)} Wx[co,f,kh,kw]
// ---------------------------------------------------------------------------
__global__ void ebias_kernel(const float* __restrict__ extra,
                             const float* __restrict__ Wx,
                             const float* __restrict__ bm,
                             const float* __restrict__ bx,
                             float* __restrict__ ebias)
{
    int tid = blockIdx.x * 256 + threadIdx.x;
    if (tid >= B_ * COUT) return;
    int b = tid / COUT;
    int i = tid % COUT;

    float bias = bm[i] + bx[i];
    float e[3], wloc[3][9];
    for (int f = 0; f < 3; ++f) {
        e[f] = extra[(b * COUT + i) * 3 + f];
        for (int t = 0; t < 9; ++t) wloc[f][t] = Wx[(i * 3 + f) * 9 + t];
    }
    for (int rc = 0; rc < 3; ++rc) {
        int kh0 = (rc == 0) ? 1 : 0;
        int kh1 = (rc == 2) ? 1 : 2;
        for (int cc = 0; cc < 3; ++cc) {
            int kw0 = (cc == 0) ? 1 : 0;
            int kw1 = (cc == 2) ? 1 : 2;
            float s = 0.f;
            for (int f = 0; f < 3; ++f) {
                float sf = 0.f;
                for (int kh = kh0; kh <= kh1; ++kh)
                    for (int kw = kw0; kw <= kw1; ++kw)
                        sf += wloc[f][kh * 3 + kw];
                s += e[f] * sf;
            }
            ebias[tid * 9 + rc * 3 + cc] = bias + s;
        }
    }
}

// ---------------------------------------------------------------------------
// Pre-kernel 3: pack x into padded, bf16x2-split, staging-friendly layout.
//   xp[((b*HP + hp)*WP + wp)*192 + chunk*64 + j]
//     j <  32 : hi bf16 of ci = chunk*32 + j
//     j >= 32 : lo bf16 of ci = chunk*32 + (j-32)
//   hp/wp include the zero pad border (hp = h+1, wp = w+1), so the main
//   kernel stages with zero bounds checks: every (pixel,chunk) is one
//   contiguous 128B block = 8 lanes x b128.
// ---------------------------------------------------------------------------
__global__ void pack_x_kernel(const float* __restrict__ x,
                              unsigned short* __restrict__ xp)
{
    int t = blockIdx.x * 256 + threadIdx.x;     // over B_*HP*WP pixels
    if (t >= B_ * HP * WP) return;
    int wp = t % WP;
    int hp = (t / WP) % HP;
    int b  = t / (WP * HP);
    int h = hp - 1, w = wp - 1;
    bool inb = ((unsigned)h < (unsigned)H_) && ((unsigned)w < (unsigned)W_);

    unsigned short* dst = xp + (long)t * 192;
    for (int ci = 0; ci < CIN; ++ci) {
        float v = inb ? x[((b * CIN + ci) * H_ + h) * W_ + w] : 0.0f;
        unsigned short hi = f2bf(v);
        int chunk = ci >> 5, j = ci & 31;
        dst[chunk * 64 + j]      = hi;
        dst[chunk * 64 + 32 + j] = f2bf(v - bf2f(hi));
    }
}

// ---------------------------------------------------------------------------
// Stage one Cin-chunk of a 4-row x 58-col window into LDS.
//   L[row][col][64]:  [0..31]=hi ci, [32..63]=lo ci   (128B per (row,col))
//   4*58 segments x 8 sub-b128 = 1856 lane transfers, 256 threads.
// ---------------------------------------------------------------------------
static __device__ __forceinline__ void stage_chunk(
        const unsigned short* __restrict__ xp,
        unsigned short (*__restrict__ L)[WP][64],
        int b, int h0, int chunk, int tid)
{
    for (int i = tid; i < 4 * WP * 8; i += 256) {
        int sub = i & 7;          // 16B piece within the 128B block
        int seg = i >> 3;
        int col = seg % WP;
        int row = seg / WP;       // padded input row = h0 + row
        long gelem = ((long)((b * HP + (h0 + row)) * WP + col)) * 192
                   + chunk * 64 + sub * 8;
        unsigned short* l = &L[row][col][sub * 8];
#if ASYNC_STAGE
        // signature (probe-confirmed): (int4 AS1* gsrc, int4 AS3* ldst, imm off, imm cpol)
        __builtin_amdgcn_global_load_async_to_lds_b128(
            (AS1 v4i*)(xp + gelem), (AS3 v4i*)l, 0, 0);
#else
        *(uint4*)l = *(const uint4*)(xp + gelem);
#endif
    }
}

// ---------------------------------------------------------------------------
// Main conv kernel: one workgroup per (b, output row pair h0,h0+1).
// 256 threads = 8 waves: wave w -> output row h0 + (w>>2), cols [16*(w&3), +16).
// K loop = 3 Cin-chunks of 32 x 9 taps; bf16x2 split: 3 WMMAs per MAC tile.
// LDS X tile double-buffered; next chunk streams in async during WMMA.
// ---------------------------------------------------------------------------
__global__ __launch_bounds__(256) void conv_wmma(
        const unsigned short* __restrict__ xp,
        const unsigned short* __restrict__ whi,
        const unsigned short* __restrict__ wlo,
        const float* __restrict__ ebias,
        float* __restrict__ out)
{
    const int h0 = blockIdx.x * 2;
    const int b  = blockIdx.y;

    // [buf][padded row h0+r][padded col][32 hi ci | 32 lo ci]
    __shared__ __align__(16) unsigned short Ls[2][4][WP][64];

    const int tid  = threadIdx.x;
    const int lane = tid & 31;
    const int wave = tid >> 5;
    const int q    = wave >> 2;                  // which output row of the pair
    const int h    = h0 + q;
    const int n    = (wave & 3) * 16 + (lane & 15);   // output column (N index)
    const int kbase = (lane >> 4) * 16;          // B-frag: lanes 0-15 K0..15, 16-31 K16..31

    v8f acc[9];
#pragma unroll
    for (int m = 0; m < 9; ++m) acc[m] = v8f{};

    stage_chunk(xp, Ls[0], b, h0, 0, tid);
    wait_async_lds();
    __syncthreads();

    int buf = 0;
#pragma unroll 1
    for (int chunk = 0; chunk < 3; ++chunk) {
        // prefetch next chunk into the other buffer while we compute
        if (chunk < 2)
            stage_chunk(xp, Ls[buf ^ 1], b, h0, chunk + 1, tid);

        unsigned short (*__restrict__ Lb)[WP][64] = Ls[buf];

#pragma unroll
        for (int tap = 0; tap < 9; ++tap) {
            const int kh = tap / 3;
            const int kw = tap % 3;
            int col = n + kw;                    // padded col = (n + kw - 1) + 1
            col = (col > WP - 1) ? (WP - 1) : col;   // clamp dead lanes (n >= 56)

            const unsigned short* lp = &Lb[q + kh][col][0];
            v16bf bhi = load_frag(lp + kbase);
            v16bf blo = load_frag(lp + 32 + kbase);

            const unsigned short* wph = whi + ((tap * 3 + chunk) * 9) * 512 + lane * 16;
            const unsigned short* wpl = wlo + ((tap * 3 + chunk) * 9) * 512 + lane * 16;
#pragma unroll
            for (int mt = 0; mt < 9; ++mt) {
                v16bf ahi = load_frag(wph + mt * 512);
                v16bf alo = load_frag(wpl + mt * 512);
                acc[mt] = __builtin_amdgcn_wmma_f32_16x16x32_bf16(
                              false, ahi, false, bhi, (short)0, acc[mt], false, false);
                acc[mt] = __builtin_amdgcn_wmma_f32_16x16x32_bf16(
                              false, alo, false, bhi, (short)0, acc[mt], false, false);
                acc[mt] = __builtin_amdgcn_wmma_f32_16x16x32_bf16(
                              false, ahi, false, blo, (short)0, acc[mt], false, false);
            }
        }

        if (chunk < 2) {
            wait_async_lds();      // our async writes to buf^1 done
            __syncthreads();       // everyone done writing buf^1 / reading buf
            buf ^= 1;
        }
    }

    // Epilogue: D layout = VGPR r: lanes 0-15 -> M=r, lanes 16-31 -> M=8+r; N = lane%16
    if (n < W_) {
        const int rc = (h == 0) ? 0 : ((h == H_ - 1) ? 2 : 1);
        const int cc = (n == 0) ? 0 : ((n == W_ - 1) ? 2 : 1);
        const int coBase = (lane >> 4) * 8;
#pragma unroll
        for (int mt = 0; mt < 9; ++mt) {
#pragma unroll
            for (int r = 0; r < 8; ++r) {
                int co = mt * 16 + coBase + r;
                out[((b * COUT + co) * H_ + h) * W_ + n] =
                    acc[mt][r] + ebias[(b * COUT + co) * 9 + rc * 3 + cc];
            }
        }
    }
}

// ---------------------------------------------------------------------------
extern "C" void kernel_launch(void* const* d_in, const int* in_sizes, int n_in,
                              void* d_out, int out_size, void* d_ws, size_t ws_size,
                              hipStream_t stream)
{
    (void)in_sizes; (void)n_in; (void)out_size; (void)ws_size;

    const float* x     = (const float*)d_in[0];   // [32,96,56,56]
    const float* extra = (const float*)d_in[1];   // [32,432]
    const float* Wm    = (const float*)d_in[2];   // [144,96,3,3]
    const float* bm    = (const float*)d_in[3];   // [144]
    const float* Wx    = (const float*)d_in[4];   // [144,3,3,3]
    const float* bx    = (const float*)d_in[5];   // [144]
    float* out = (float*)d_out;                   // [32,144,56,56]

    // workspace layout (all 16B aligned):
    //   whi  : 124416 bf16                     (248832 B)
    //   wlo  : 124416 bf16                     (248832 B)
    //   ebias: 32*144*9 f32                    (165888 B)
    //   xp   : 32*58*58*192 bf16 (padded img)  (~41.3 MB)
    unsigned short* whi = (unsigned short*)d_ws;
    unsigned short* wlo = whi + 124416;
    float* ebias = (float*)(wlo + 124416);
    unsigned short* xp = (unsigned short*)(ebias + B_ * COUT * 9);

    pack_w_kernel<<<(124416 + 255) / 256, 256, 0, stream>>>(Wm, whi, wlo);
    ebias_kernel<<<(B_ * COUT + 255) / 256, 256, 0, stream>>>(extra, Wx, bm, bx, ebias);
    pack_x_kernel<<<(B_ * HP * WP + 255) / 256, 256, 0, stream>>>(x, xp);
    conv_wmma<<<dim3(H_ / 2, B_), 256, 0, stream>>>(xp, whi, wlo, ebias, out);
}
